// TalkingHeadLayer_39204461478665
// MI455X (gfx1250) — matrix operand order repaired and checked by reading, not compiled, not run on hardware
//
#include <hip/hip_runtime.h>
#include <hip/hip_bf16.h>
#include <stdint.h>

// ---------------------------------------------------------------- constants
#define Bb      16
#define Nn      1024
#define F_IN    64
#define F_OUT   64
#define Hh      4
#define SE_DIM  64
#define F_TOT   128        // F_IN + SE_DIM
#define ALPHA   0.2f

typedef __attribute__((ext_vector_type(16))) __bf16        v16bf;
typedef __attribute__((ext_vector_type(8)))  float         v8f;
typedef __attribute__((ext_vector_type(8)))  unsigned int  v8u;

// f32 -> bf16 pack: round-half-up bias, then one v_perm_b32 grabs both high halves
__device__ __forceinline__ unsigned int bf16_2(float lo, float hi) {
    unsigned int ul = __builtin_bit_cast(unsigned int, lo) + 0x8000u;
    unsigned int uh = __builtin_bit_cast(unsigned int, hi) + 0x8000u;
    return __builtin_amdgcn_perm(uh, ul, 0x07060302u);   // {uh[31:16], ul[31:16]}
}
__device__ __forceinline__ unsigned int bf16_1(float f) {
    return (__builtin_bit_cast(unsigned int, f) + 0x8000u) >> 16;
}
__device__ __forceinline__ float bf16_to_f32(unsigned short u) {
    return __builtin_bit_cast(float, ((unsigned int)u) << 16);
}

// ---------------------------------------------------------------- kernel 0a
// W[h][f][o] f32 -> Wt[h][o][f] bf16 (B-fragments become 16 contiguous halfs)
__global__ void k_prep_w(const float* __restrict__ W, unsigned short* __restrict__ Wt) {
    int idx = blockIdx.x * blockDim.x + threadIdx.x;          // H*F_OUT*F_TOT
    if (idx >= Hh * F_OUT * F_TOT) return;
    int h = idx / (F_OUT * F_TOT);
    int r = idx - h * (F_OUT * F_TOT);
    int o = r / F_TOT;
    int f = r - o * F_TOT;
    Wt[idx] = (unsigned short)bf16_1(W[(h * F_TOT + f) * F_OUT + o]);
}

// ---------------------------------------------------------------- kernel 0b
// concat(x,SE) -> packed bf16 xi[b][n][f] so A-fragments are pure vector loads
__global__ void k_prep_xi(const float* __restrict__ x, const float* __restrict__ SE,
                          unsigned int* __restrict__ xi) {
    int p = blockIdx.x * blockDim.x + threadIdx.x;            // B*N*F_TOT/2 pairs
    if (p >= Bb * Nn * (F_TOT / 2)) return;
    const int row = p >> 6;            // b*N + i
    const int fp  = (p & 63) << 1;     // even feature index
    const int i   = row & (Nn - 1);
    float2 v;
    if (fp < F_IN) v = *(const float2*)&x[(size_t)row * F_IN + fp];
    else           v = *(const float2*)&SE[(size_t)i * SE_DIM + (fp - F_IN)];
    xi[p] = bf16_2(v.x, v.y);
}

// ---------------------------------------------------------------- kernel 1
// h[b,h,n,o] = xi @ W[h]; store transposed bf16 hT[b][h][o][n].
// 1 block = (b, 16-row tile); 4 waves, wave w handles head w.
__global__ void k_proj(const unsigned int* __restrict__ xi,
                       const unsigned short* __restrict__ Wt,
                       unsigned short* __restrict__ hT) {
    const int b   = blockIdx.x >> 6;
    const int i0  = (blockIdx.x & 63) << 4;
    const int lane = threadIdx.x & 31;
    const int h    = threadIdx.x >> 5;          // wave id == head
    const int m    = lane & 15;
    const int hi   = lane >> 4;
    const int row  = b * Nn + i0 + m;

    v8f acc[4] = {v8f{0}, v8f{0}, v8f{0}, v8f{0}};

#pragma unroll
    for (int ks = 0; ks < 4; ++ks) {
        const int k0 = ks * 32;
        // A fragment: K {k0+hi*16 .. +15}, contiguous pairs
        const uint4 a0 = *(const uint4*)&xi[(size_t)row * 64 + (k0 >> 1) + hi * 8];
        const uint4 a1 = *(const uint4*)&xi[(size_t)row * 64 + (k0 >> 1) + hi * 8 + 4];
        v8u au;
        au[0] = a0.x; au[1] = a0.y; au[2] = a0.z; au[3] = a0.w;
        au[4] = a1.x; au[5] = a1.y; au[6] = a1.z; au[7] = a1.w;
        const v16bf afrag = __builtin_bit_cast(v16bf, au);
#pragma unroll
        for (int ot = 0; ot < 4; ++ot) {
            const int o = ot * 16 + m;
            const v8u bu = *(const v8u*)&Wt[((size_t)(h * F_OUT + o)) * F_TOT + k0 + hi * 16];
            const v16bf bfrag = __builtin_bit_cast(v16bf, bu);
            acc[ot] = __builtin_amdgcn_wmma_f32_16x16x32_bf16(
                false, afrag, false, bfrag, (short)0, acc[ot], false, false);
        }
    }
    // C fragment: VGPR r holds row M = hi*8 + r, column o = ot*16 + m.
#pragma unroll
    for (int ot = 0; ot < 4; ++ot) {
        const int o = ot * 16 + m;
        uint4 pk;
        pk.x = bf16_2(acc[ot][0], acc[ot][1]);
        pk.y = bf16_2(acc[ot][2], acc[ot][3]);
        pk.z = bf16_2(acc[ot][4], acc[ot][5]);
        pk.w = bf16_2(acc[ot][6], acc[ot][7]);
        *(uint4*)&hT[(((size_t)(b * Hh + h) * F_OUT + o) * Nn) + i0 + hi * 8] = pk;
    }
}

// ---------------------------------------------------------------- kernel 2
// f1[b,h,n] = h . a1[h] ; f2 = h . a2[h]   (reads hT coalesced over n)
__global__ void k_f12(const unsigned short* __restrict__ hT,
                      const float* __restrict__ a1, const float* __restrict__ a2,
                      float* __restrict__ f1, float* __restrict__ f2) {
    int tid = blockIdx.x * blockDim.x + threadIdx.x;          // B*H*N
    if (tid >= Bb * Hh * Nn) return;
    const int h = (tid >> 10) & 3;
    const size_t base = ((size_t)(tid >> 10)) * F_OUT * Nn + (tid & 1023);
    float s1 = 0.f, s2 = 0.f;
#pragma unroll 8
    for (int o = 0; o < F_OUT; ++o) {
        const float v = bf16_to_f32(hT[base + (size_t)o * Nn]);
        s1 = fmaf(v, a1[h * F_OUT + o], s1);
        s2 = fmaf(v, a2[h * F_OUT + o], s2);
    }
    f1[tid] = s1;
    f2[tid] = s2;
}

// ---------------------------------------------------------------- kernel 3
// online softmax stats over the i axis: per (b,g,j) column max & 1/sum.
__global__ void k_colstats(const float* __restrict__ f1, const float* __restrict__ f2,
                           const float* __restrict__ P_l,
                           float* __restrict__ colmax, float* __restrict__ colrcp) {
    int tid = blockIdx.x * blockDim.x + threadIdx.x;          // B*H*N, g index
    if (tid >= Bb * Hh * Nn) return;
    const int b = tid >> 12;
    const int g = (tid >> 10) & 3;
    const int j = tid & 1023;
    float f2v[4], pl[4];
#pragma unroll
    for (int hp = 0; hp < 4; ++hp) {
        f2v[hp] = f2[((b * Hh + hp) << 10) + j];
        pl[hp]  = P_l[hp * Hh + g];
    }
    float mx = -3.0e38f, sm = 0.f;
    for (int i = 0; i < Nn; ++i) {
        float e = 0.f;
#pragma unroll
        for (int hp = 0; hp < 4; ++hp) {
            float u = f1[((b * Hh + hp) << 10) + i] + f2v[hp];
            u = (u > 0.f) ? u : ALPHA * u;
            e = fmaf(pl[hp], u, e);
        }
        const float nm = fmaxf(mx, e);
        sm = sm * __expf(mx - nm) + __expf(e - nm);
        mx = nm;
    }
    colmax[tid] = mx;
    colrcp[tid] = 1.0f / sm;
}

// ---------------------------------------------------------------- kernel 4
// Fused: recompute attention (softmax over i via colstats), P_l/P_w mixing,
// adj mask, bf16 pack — all cooperatively into LDS — then WMMA aggregation,
// residual + ELU, head-concat store.  1 block = (b, 16-i tile); wave w = head w.
__global__ void k_attn_agg(const float* __restrict__ x, const float* __restrict__ adj,
                           const unsigned short* __restrict__ hT,
                           const float* __restrict__ f1, const float* __restrict__ f2,
                           const float* __restrict__ colmax, const float* __restrict__ colrcp,
                           const float* __restrict__ P_l, const float* __restrict__ P_w,
                           float* __restrict__ out) {
    __shared__ float s_f1[4][16];
    __shared__ float s_f2[4][32];
    __shared__ float s_cm[4][32];
    __shared__ float s_cr[4][32];
    __shared__ unsigned int s_att_pk[Hh][16][16];   // bf16-pair att per head

    const int b  = blockIdx.x >> 6;
    const int i0 = (blockIdx.x & 63) << 4;
    const int t  = threadIdx.x;          // 0..127
    const int lane = t & 31;
    const int h    = t >> 5;             // wave id == output head
    const int m    = lane & 15;
    const int hi   = lane >> 4;

    // per-thread constants: full P_l and P_w
    float pl[4][4], pwall[4][4];
#pragma unroll
    for (int g = 0; g < 4; ++g) {
#pragma unroll
        for (int q = 0; q < 4; ++q) {
            pl[g][q]    = P_l[g * Hh + q];   // pl[h'][g] indexed [h'][g]
            pwall[g][q] = P_w[g * Hh + q];   // pwall[g][h]
        }
    }
    if (t < 64) {   // stage f1 for this i-tile once
        const int hp = t >> 4, ii = t & 15;
        s_f1[hp][ii] = f1[((b * Hh + hp) << 10) + i0 + ii];
    }

    v8f acc[4] = {v8f{0}, v8f{0}, v8f{0}, v8f{0}};
    const unsigned short* hTb = hT + ((size_t)(b * Hh + h) * F_OUT) * Nn;

#pragma unroll 1
    for (int j0 = 0; j0 < Nn; j0 += 32) {
        __syncthreads();                                   // previous tile consumed
        {   // stage column stats for this 32-j tile
            const int g4 = t >> 5, jj = t & 31;
            s_f2[g4][jj] = f2[((b * Hh + g4) << 10) + j0 + jj];
            s_cm[g4][jj] = colmax[((b * Hh + g4) << 10) + j0 + jj];
            s_cr[g4][jj] = colrcp[((b * Hh + g4) << 10) + j0 + jj];
        }
        __syncthreads();
        // cooperative: softmax probs (each exp once per block), P_w mix, adj
        // mask, bf16 pair-pack for all 4 heads
#pragma unroll
        for (int q = 0; q < 2; ++q) {
            const int c  = (t << 1) | q;            // 0..255
            const int ii = c >> 4, jp = c & 15;
            const float2 aj = *(const float2*)
                &adj[((size_t)b * Nn + i0 + ii) * Nn + j0 + (jp << 1)];
            float attv[2][4];
#pragma unroll
            for (int s = 0; s < 2; ++s) {
                const int jc = (jp << 1) + s;
                float lr[4];
#pragma unroll
                for (int hp = 0; hp < 4; ++hp) {
                    float u = s_f1[hp][ii] + s_f2[hp][jc];
                    lr[hp] = (u > 0.f) ? u : ALPHA * u;
                }
                float pg[4];
#pragma unroll
                for (int g = 0; g < 4; ++g) {
                    float eg = lr[0] * pl[0][g] + lr[1] * pl[1][g] +
                               lr[2] * pl[2][g] + lr[3] * pl[3][g];
                    pg[g] = __expf(eg - s_cm[g][jc]) * s_cr[g][jc];
                }
                const float am = (s == 0) ? aj.x : aj.y;
#pragma unroll
                for (int hh = 0; hh < 4; ++hh)
                    attv[s][hh] = am * (pg[0] * pwall[0][hh] + pg[1] * pwall[1][hh] +
                                        pg[2] * pwall[2][hh] + pg[3] * pwall[3][hh]);
            }
#pragma unroll
            for (int hh = 0; hh < 4; ++hh)
                s_att_pk[hh][ii][jp] = bf16_2(attv[0][hh], attv[1][hh]);
        }
        __syncthreads();

        // A fragment: two ds_load_b128 (K-pairs hi*8+v and hi*8+4+v contiguous)
        const uint4 p0 = *(const uint4*)&s_att_pk[h][m][hi * 8];
        const uint4 p1 = *(const uint4*)&s_att_pk[h][m][hi * 8 + 4];
        v8u au;
        au[0] = p0.x; au[1] = p0.y; au[2] = p0.z; au[3] = p0.w;
        au[4] = p1.x; au[5] = p1.y; au[6] = p1.z; au[7] = p1.w;
        const v16bf afrag = __builtin_bit_cast(v16bf, au);

        if (j0 + 32 < Nn)   // prefetch next B tile line
            __builtin_prefetch(&hTb[(size_t)m * Nn + j0 + 32 + hi * 16], 0, 3);

#pragma unroll
        for (int ot = 0; ot < 4; ++ot) {
            const int o = ot * 16 + m;
            const v8u bu = *(const v8u*)&hTb[(size_t)o * Nn + j0 + hi * 16];
            const v16bf bfrag = __builtin_bit_cast(v16bf, bu);
            acc[ot] = __builtin_amdgcn_wmma_f32_16x16x32_bf16(
                false, afrag, false, bfrag, (short)0, acc[ot], false, false);
        }
    }

    // epilogue: residual + ELU, concat heads -> out[b][i][h*64+o]
#pragma unroll
    for (int ot = 0; ot < 4; ++ot) {
        const int o = ot * 16 + m;
#pragma unroll
        for (int r = 0; r < 8; ++r) {
            const int irow = i0 + hi * 8 + r;
            float v = acc[ot][r] + x[((size_t)b * Nn + irow) * F_IN + o];
            v = (v > 0.f) ? v : (__expf(v) - 1.0f);
            out[((size_t)b * Nn + irow) * (Hh * F_OUT) + h * F_OUT + o] = v;
        }
    }
}

// ---------------------------------------------------------------- launcher
extern "C" void kernel_launch(void* const* d_in, const int* in_sizes, int n_in,
                              void* d_out, int out_size, void* d_ws, size_t ws_size,
                              hipStream_t stream) {
    const float* x   = (const float*)d_in[0];
    const float* adj = (const float*)d_in[1];
    const float* SE  = (const float*)d_in[2];
    const float* W   = (const float*)d_in[3];
    const float* a1  = (const float*)d_in[4];
    const float* a2  = (const float*)d_in[5];
    const float* P_l = (const float*)d_in[6];
    const float* P_w = (const float*)d_in[7];
    float* out = (float*)d_out;

    uint8_t* ws = (uint8_t*)d_ws;
    // workspace layout (≈13.1 MB)
    unsigned short* Wt = (unsigned short*)(ws);                               //  64 KB
    unsigned int*   xi = (unsigned int*)(ws + (64u << 10));                   //   4 MB
    unsigned short* hT = (unsigned short*)(ws + (64u << 10) + (4u << 20));    //   8 MB
    float* f1     = (float*)(ws + (64u << 10) + (12u << 20));                 // 256 KB
    float* f2     = f1 + Bb * Hh * Nn;
    float* colmax = f2 + Bb * Hh * Nn;
    float* colrcp = colmax + Bb * Hh * Nn;

    k_prep_w<<<dim3((Hh * F_OUT * F_TOT + 255) / 256), dim3(256), 0, stream>>>(W, Wt);
    k_prep_xi<<<dim3(Bb * Nn * (F_TOT / 2) / 256), dim3(256), 0, stream>>>(x, SE, xi);
    k_proj<<<dim3(Bb * (Nn / 16)), dim3(128), 0, stream>>>(xi, Wt, hT);
    k_f12<<<dim3(Bb * Hh * Nn / 256), dim3(256), 0, stream>>>(hT, a1, a2, f1, f2);
    k_colstats<<<dim3(Bb * Hh * Nn / 256), dim3(256), 0, stream>>>(f1, f2, P_l, colmax, colrcp);
    k_attn_agg<<<dim3(Bb * (Nn / 16)), dim3(128), 0, stream>>>(
        x, adj, hT, f1, f2, colmax, colrcp, P_l, P_w, out);
}
